// Encoder_41936060678647
// MI455X (gfx1250) — compile-verified, matrix operand order checked by reading, not run on hardware
//
#include <hip/hip_runtime.h>
#include <math.h>

// ---------------------------------------------------------------- types
typedef __bf16 bf16_t;
typedef __attribute__((ext_vector_type(16))) __bf16 v16bf;
typedef __attribute__((ext_vector_type(8)))  __bf16 v8bf;
typedef __attribute__((ext_vector_type(8)))  float  v8f;
typedef __attribute__((ext_vector_type(4)))  int    int4v;

#define LAYERS 6
#define DMODEL 1024
#define FFDIM  4096
#define NHEAD  16
#define DHEAD  64
#define SEQ    2048
#define BATCH  4
#define NTOK   (BATCH*SEQ)

// Async global->LDS copy support (CDNA5, tracked by ASYNCcnt)
#if defined(__has_builtin)
#  if __has_builtin(__builtin_amdgcn_global_load_async_to_lds_b128)
#    define USE_ASYNC_LDS 1
#  endif
#endif
#ifndef USE_ASYNC_LDS
#  define USE_ASYNC_LDS 0
#endif

typedef __attribute__((address_space(1))) int4v* gint4_p;  // global int4*
typedef __attribute__((address_space(3))) int4v* lint4_p;  // LDS int4*

static __device__ inline void copy16_g2l(const bf16_t* g, bf16_t* l) {
#if USE_ASYNC_LDS
  __builtin_amdgcn_global_load_async_to_lds_b128((gint4_p)g, (lint4_p)l, 0, 0);
#else
  *(uint4*)l = *(const uint4*)g;
#endif
}

static __device__ inline void wait_async_copies() {
#if USE_ASYNC_LDS
  asm volatile("s_wait_asynccnt 0" ::: "memory");
#endif
}

// Build a 16xK-slice bf16 fragment from two 8-element halves
// (CDNA5 16-bit A/B VGPR layout: per lane, 8 bf16 at kbase, 8 at kbase+16)
static __device__ inline v16bf ld_frag16(const bf16_t* p0, const bf16_t* p1) {
  union { v16bf v; v8bf h[2]; } u;
  u.h[0] = *(const v8bf*)p0;
  u.h[1] = *(const v8bf*)p1;
  return u.v;
}

static __device__ inline v8f wmma_bf16(v16bf a, v16bf b, v8f c) {
  return __builtin_amdgcn_wmma_f32_16x16x32_bf16(
      false, a, false, b, (short)0, c, false, false);
}

// ---------------------------------------------------------------- GEMM
// C[M,N] = act(A[M,K] @ Bt[N,K]^T + bias).  Block tile 256x128 (8 waves,
// wave tile 64x64 = 16 WMMAs/k-step), BK=32, double-buffered LDS with
// async global->LDS copies.
template<bool RELU, bool OUTF, bool OUTB>
__global__ __launch_bounds__(256) void gemm_bf16_wmma(
    const bf16_t* __restrict__ A,    // [M,K] row-major
    const bf16_t* __restrict__ Bt,   // [N,K] row-major (pre-transposed weights)
    const float*  __restrict__ bias, // [N]
    float*        __restrict__ Cf,   // fp32 out (used if OUTF)
    bf16_t*       __restrict__ Cb,   // bf16 out (used if OUTB)
    int M, int N, int K)
{
  __shared__ __align__(16) bf16_t As[2][256][32];
  __shared__ __align__(16) bf16_t Bs[2][128][32];

  const int tid  = threadIdx.x;
  const int wave = tid >> 5;
  const int lane = tid & 31;
  const int wm   = wave >> 1;        // 0..3 : 64-row slab
  const int wn   = wave & 1;         // 0..1 : 64-col slab
  const int bm   = blockIdx.y * 256;
  const int bn   = blockIdx.x * 128;
  const int hi   = lane >> 4;        // K-half select (wave32 halves)
  const int lr   = lane & 15;

  v8f acc[4][4] = {};

  auto load_tiles = [&](int buf, int k0) {
    // A tile: 256x32 bf16 = 1024 x 16B chunks, 4/thread
#pragma unroll
    for (int c = 0; c < 4; ++c) {
      int idx = tid + c * 256;
      int r = idx >> 2, kc = (idx & 3) << 3;
      copy16_g2l(&A[(size_t)(bm + r) * K + k0 + kc], &As[buf][r][kc]);
    }
    // B tile: 128x32 bf16 = 512 x 16B chunks, 2/thread
#pragma unroll
    for (int c = 0; c < 2; ++c) {
      int idx = tid + c * 256;
      int r = idx >> 2, kc = (idx & 3) << 3;
      copy16_g2l(&Bt[(size_t)(bn + r) * K + k0 + kc], &Bs[buf][r][kc]);
    }
  };

  int buf = 0;
  load_tiles(0, 0);
  wait_async_copies();
  __syncthreads();

  for (int k0 = 0; k0 < K; k0 += 32) {
    if (k0 + 32 < K) load_tiles(buf ^ 1, k0 + 32);

    v16bf af[4], bfv[4];
#pragma unroll
    for (int mi = 0; mi < 4; ++mi) {
      const bf16_t* rp = &As[buf][wm * 64 + mi * 16 + lr][0];
      af[mi] = ld_frag16(rp + hi * 8, rp + 16 + hi * 8);
    }
#pragma unroll
    for (int ni = 0; ni < 4; ++ni) {
      const bf16_t* rp = &Bs[buf][wn * 64 + ni * 16 + lr][0];
      bfv[ni] = ld_frag16(rp + hi * 8, rp + 16 + hi * 8);
    }
#pragma unroll
    for (int mi = 0; mi < 4; ++mi)
#pragma unroll
      for (int ni = 0; ni < 4; ++ni)
        acc[mi][ni] = wmma_bf16(af[mi], bfv[ni], acc[mi][ni]);

    wait_async_copies();
    __syncthreads();
    buf ^= 1;
  }

  // epilogue: C layout => VGPR r : lanes 0-15 -> M=r, lanes 16-31 -> M=r+8
#pragma unroll
  for (int mi = 0; mi < 4; ++mi)
#pragma unroll
    for (int ni = 0; ni < 4; ++ni) {
      int col = bn + wn * 64 + ni * 16 + lr;
      float bv = bias[col];
#pragma unroll
      for (int r = 0; r < 8; ++r) {
        int row = bm + wm * 64 + mi * 16 + hi * 8 + r;
        float v = acc[mi][ni][r] + bv;
        if (RELU) v = fmaxf(v, 0.0f);
        if (OUTF) Cf[(size_t)row * N + col] = v;
        if (OUTB) Cb[(size_t)row * N + col] = (bf16_t)v;
      }
    }
}

// ---------------------------------------------------------------- attention
// One wave handles 16 queries of one (b,h); loop over keys in chunks of 32.
// Q,K: bf16 [B,S,H,DH]; Vt: bf16 [B,H,DH,S]; O: bf16 [B,S,D].
__global__ __launch_bounds__(256) void attention_wmma(
    const bf16_t* __restrict__ Q,
    const bf16_t* __restrict__ Km,
    const bf16_t* __restrict__ Vt,
    bf16_t*       __restrict__ O)
{
  __shared__ __align__(16) bf16_t Pls[8][16][32];

  const int wave = threadIdx.x >> 5;
  const int lane = threadIdx.x & 31;
  const int hi = lane >> 4, lr = lane & 15;
  const int h  = blockIdx.y, b = blockIdx.z;
  const int q0 = blockIdx.x * 128 + wave * 16;

  const bf16_t* qrow = Q + ((size_t)(b * SEQ + q0 + lr) * NHEAD + h) * DHEAD;
  v16bf qf[2];
#pragma unroll
  for (int c = 0; c < 2; ++c)
    qf[c] = ld_frag16(qrow + c * 32 + hi * 8, qrow + c * 32 + 16 + hi * 8);

  v8f accO[4] = {};
  float mrow[8], lrow[8];
#pragma unroll
  for (int r = 0; r < 8; ++r) { mrow[r] = -1e30f; lrow[r] = 0.0f; }

  const float scale = 0.125f;  // 1/sqrt(64)

  for (int kt0 = 0; kt0 < SEQ; kt0 += 32) {
    v8f s[2];
#pragma unroll
    for (int j = 0; j < 2; ++j) {
      const bf16_t* krow =
          Km + ((size_t)(b * SEQ + kt0 + j * 16 + lr) * NHEAD + h) * DHEAD;
      v16bf kf0 = ld_frag16(krow + hi * 8,      krow + 16 + hi * 8);
      v16bf kf1 = ld_frag16(krow + 32 + hi * 8, krow + 48 + hi * 8);
      v8f z = {};
      z = wmma_bf16(qf[0], kf0, z);
      z = wmma_bf16(qf[1], kf1, z);
      s[j] = z;
    }

    // online softmax; C-layout element r == (query hi*8+r, key lr)
    float p[2][8];
#pragma unroll
    for (int r = 0; r < 8; ++r) {
      float s0 = s[0][r] * scale, s1 = s[1][r] * scale;
      float mx = fmaxf(s0, s1);
      for (int msk = 1; msk < 16; msk <<= 1)
        mx = fmaxf(mx, __shfl_xor(mx, msk, 32));
      float mnew  = fmaxf(mrow[r], mx);
      float alpha = __expf(mrow[r] - mnew);
      float p0 = __expf(s0 - mnew), p1 = __expf(s1 - mnew);
      float rs = p0 + p1;
      for (int msk = 1; msk < 16; msk <<= 1)
        rs += __shfl_xor(rs, msk, 32);
      lrow[r] = lrow[r] * alpha + rs;
      mrow[r] = mnew;
      p[0][r] = p0; p[1][r] = p1;
#pragma unroll
      for (int ni = 0; ni < 4; ++ni) accO[ni][r] *= alpha;
    }

    // re-layout P (C layout -> A layout) through LDS
    __syncthreads();
#pragma unroll
    for (int r = 0; r < 8; ++r) {
      Pls[wave][hi * 8 + r][lr]      = (bf16_t)p[0][r];
      Pls[wave][hi * 8 + r][16 + lr] = (bf16_t)p[1][r];
    }
    __syncthreads();
    v16bf pf = ld_frag16(&Pls[wave][lr][hi * 8], &Pls[wave][lr][16 + hi * 8]);

#pragma unroll
    for (int ni = 0; ni < 4; ++ni) {
      const bf16_t* vrow =
          Vt + ((size_t)(b * NHEAD + h) * DHEAD + ni * 16 + lr) * SEQ + kt0;
      v16bf vf = ld_frag16(vrow + hi * 8, vrow + 16 + hi * 8);
      accO[ni] = wmma_bf16(pf, vf, accO[ni]);
    }
  }

#pragma unroll
  for (int ni = 0; ni < 4; ++ni)
#pragma unroll
    for (int r = 0; r < 8; ++r) {
      int q  = q0 + hi * 8 + r;
      int dh = ni * 16 + lr;
      O[(size_t)(b * SEQ + q) * DMODEL + h * DHEAD + dh] =
          (bf16_t)(accO[ni][r] / lrow[r]);
    }
}

// ---------------------------------------------------------------- residual + LN
__global__ __launch_bounds__(256) void add_layernorm(
    const float* __restrict__ X, const float* __restrict__ Fv,
    float* __restrict__ Of, bf16_t* __restrict__ Ob)
{
  __shared__ float part[8], part2[8];
  const int row = blockIdx.x, tid = threadIdx.x;
  const int wave = tid >> 5, lane = tid & 31;
  const float* xr = X  + (size_t)row * DMODEL;
  const float* fr = Fv + (size_t)row * DMODEL;

  float v[4]; float s = 0.0f;
#pragma unroll
  for (int i = 0; i < 4; ++i) {
    int idx = tid + i * 256;
    v[i] = xr[idx] + fr[idx];
    s += v[i];
  }
  for (int m = 1; m < 32; m <<= 1) s += __shfl_xor(s, m, 32);
  if (lane == 0) part[wave] = s;
  __syncthreads();
  float tot = 0.0f;
#pragma unroll
  for (int w = 0; w < 8; ++w) tot += part[w];
  float mean = tot * (1.0f / DMODEL);

  float vs = 0.0f;
#pragma unroll
  for (int i = 0; i < 4; ++i) { float d = v[i] - mean; vs += d * d; }
  for (int m = 1; m < 32; m <<= 1) vs += __shfl_xor(vs, m, 32);
  if (lane == 0) part2[wave] = vs;
  __syncthreads();
  float vtot = 0.0f;
#pragma unroll
  for (int w = 0; w < 8; ++w) vtot += part2[w];
  float inv = 1.0f / (sqrtf(vtot * (1.0f / DMODEL)) + 1e-6f);

#pragma unroll
  for (int i = 0; i < 4; ++i) {
    int idx = tid + i * 256;
    float o = (v[i] - mean) * inv;
    Of[(size_t)row * DMODEL + idx] = o;
    Ob[(size_t)row * DMODEL + idx] = (bf16_t)o;
  }
}

// ---------------------------------------------------------------- small utils
__global__ void cvt_f32_bf16(const float* __restrict__ in,
                             bf16_t* __restrict__ out, long long n)
{
  long long i = (long long)blockIdx.x * blockDim.x + threadIdx.x;
  if (i < n) out[i] = (bf16_t)in[i];
}

// W[K,N] fp32 -> Wt[N,K] bf16 (32x32 LDS tiles, coalesced both sides)
__global__ __launch_bounds__(256) void transpose_cvt(
    const float* __restrict__ W, bf16_t* __restrict__ Wt, int K, int N)
{
  __shared__ float t[32][33];
  const int k0 = blockIdx.y * 32, n0 = blockIdx.x * 32;
  const int tx = threadIdx.x & 31, ty = threadIdx.x >> 5;
#pragma unroll
  for (int i = ty; i < 32; i += 8)
    t[i][tx] = W[(size_t)(k0 + i) * N + n0 + tx];
  __syncthreads();
#pragma unroll
  for (int i = ty; i < 32; i += 8)
    Wt[(size_t)(n0 + i) * K + k0 + tx] = (bf16_t)t[tx][i];
}

// V [B,S,H,DH] bf16 -> Vt [B,H,DH,S] bf16
__global__ __launch_bounds__(256) void transpose_v(
    const bf16_t* __restrict__ V, bf16_t* __restrict__ Vt)
{
  __shared__ bf16_t t[32][33];
  const int bh = blockIdx.z;
  const int b = bh >> 4, h = bh & 15;
  const int s0 = blockIdx.x * 32, d0 = blockIdx.y * 32;
  const int tx = threadIdx.x & 31, ty = threadIdx.x >> 5;
#pragma unroll
  for (int i = ty; i < 32; i += 8)
    t[i][tx] = V[((size_t)(b * SEQ + s0 + i) * NHEAD + h) * DHEAD + d0 + tx];
  __syncthreads();
#pragma unroll
  for (int i = ty; i < 32; i += 8)
    Vt[((size_t)(b * NHEAD + h) * DHEAD + d0 + i) * SEQ + s0 + tx] = t[tx][i];
}

// ---------------------------------------------------------------- launcher
extern "C" void kernel_launch(void* const* d_in, const int* in_sizes, int n_in,
                              void* d_out, int out_size, void* d_ws, size_t ws_size,
                              hipStream_t stream)
{
  (void)in_sizes; (void)n_in; (void)out_size; (void)ws_size;

  const float* x_in = (const float*)d_in[0];
  const float* Wq = (const float*)d_in[1];
  const float* bq = (const float*)d_in[2];
  const float* Wk = (const float*)d_in[3];
  const float* bk = (const float*)d_in[4];
  const float* Wv = (const float*)d_in[5];
  const float* bv = (const float*)d_in[6];
  const float* Wo = (const float*)d_in[7];
  const float* bo = (const float*)d_in[8];
  const float* W1 = (const float*)d_in[9];
  const float* b1 = (const float*)d_in[10];
  const float* W2 = (const float*)d_in[11];
  const float* b2 = (const float*)d_in[12];
  float* out = (float*)d_out;

  char* ws = (char*)d_ws;
  size_t off = 0;
  auto carve = [&](size_t bytes) -> char* {
    char* p = ws + off;
    off += (bytes + 255) & ~(size_t)255;
    return p;
  };
  float*  xf   = (float*) carve((size_t)NTOK * DMODEL * 4);
  float*  tmpf = (float*) carve((size_t)NTOK * DMODEL * 4);
  bf16_t* xb   = (bf16_t*)carve((size_t)NTOK * DMODEL * 2);
  bf16_t* qb   = (bf16_t*)carve((size_t)NTOK * DMODEL * 2);
  bf16_t* kb   = (bf16_t*)carve((size_t)NTOK * DMODEL * 2);
  bf16_t* vb   = (bf16_t*)carve((size_t)NTOK * DMODEL * 2);
  bf16_t* vtb  = (bf16_t*)carve((size_t)NTOK * DMODEL * 2);
  bf16_t* ob   = (bf16_t*)carve((size_t)NTOK * DMODEL * 2);
  bf16_t* hb   = (bf16_t*)carve((size_t)NTOK * FFDIM * 2);
  bf16_t* wqt  = (bf16_t*)carve((size_t)DMODEL * DMODEL * 2);
  bf16_t* wkt  = (bf16_t*)carve((size_t)DMODEL * DMODEL * 2);
  bf16_t* wvt  = (bf16_t*)carve((size_t)DMODEL * DMODEL * 2);
  bf16_t* wot  = (bf16_t*)carve((size_t)DMODEL * DMODEL * 2);
  bf16_t* w1t  = (bf16_t*)carve((size_t)DMODEL * FFDIM * 2);
  bf16_t* w2t  = (bf16_t*)carve((size_t)FFDIM * DMODEL * 2);

  dim3 blk(256);
  long long nx = (long long)NTOK * DMODEL;
  cvt_f32_bf16<<<(unsigned)((nx + 255) / 256), blk, 0, stream>>>(x_in, xb, nx);

  const float* curx = x_in;
  const dim3 gD(DMODEL / 128, NTOK / 256);  // N=1024 GEMMs
  const dim3 gF(FFDIM  / 128, NTOK / 256);  // N=4096 GEMM

  for (int i = 0; i < LAYERS; ++i) {
    size_t wdd = (size_t)i * DMODEL * DMODEL;
    size_t wdf = (size_t)i * DMODEL * FFDIM;

    transpose_cvt<<<dim3(DMODEL/32, DMODEL/32), blk, 0, stream>>>(Wq + wdd, wqt, DMODEL, DMODEL);
    transpose_cvt<<<dim3(DMODEL/32, DMODEL/32), blk, 0, stream>>>(Wk + wdd, wkt, DMODEL, DMODEL);
    transpose_cvt<<<dim3(DMODEL/32, DMODEL/32), blk, 0, stream>>>(Wv + wdd, wvt, DMODEL, DMODEL);
    transpose_cvt<<<dim3(DMODEL/32, DMODEL/32), blk, 0, stream>>>(Wo + wdd, wot, DMODEL, DMODEL);
    transpose_cvt<<<dim3(FFDIM/32,  DMODEL/32), blk, 0, stream>>>(W1 + wdf, w1t, DMODEL, FFDIM);
    transpose_cvt<<<dim3(DMODEL/32, FFDIM/32),  blk, 0, stream>>>(W2 + wdf, w2t, FFDIM, DMODEL);

    // q/k/v = relu(x @ W + b), bf16 outputs
    gemm_bf16_wmma<true,false,true><<<gD, blk, 0, stream>>>(
        xb, wqt, bq + (size_t)i * DMODEL, nullptr, qb, NTOK, DMODEL, DMODEL);
    gemm_bf16_wmma<true,false,true><<<gD, blk, 0, stream>>>(
        xb, wkt, bk + (size_t)i * DMODEL, nullptr, kb, NTOK, DMODEL, DMODEL);
    gemm_bf16_wmma<true,false,true><<<gD, blk, 0, stream>>>(
        xb, wvt, bv + (size_t)i * DMODEL, nullptr, vb, NTOK, DMODEL, DMODEL);

    transpose_v<<<dim3(SEQ/32, DHEAD/32, BATCH*NHEAD), blk, 0, stream>>>(vb, vtb);

    attention_wmma<<<dim3(SEQ/128, NHEAD, BATCH), blk, 0, stream>>>(qb, kb, vtb, ob);

    // o = relu(attn_out @ Wo + bo), fp32
    gemm_bf16_wmma<true,true,false><<<gD, blk, 0, stream>>>(
        ob, wot, bo + (size_t)i * DMODEL, tmpf, nullptr, NTOK, DMODEL, DMODEL);
    add_layernorm<<<NTOK, blk, 0, stream>>>(curx, tmpf, xf, xb);
    curx = xf;

    // FFN
    gemm_bf16_wmma<true,false,true><<<gF, blk, 0, stream>>>(
        xb, w1t, b1 + (size_t)i * FFDIM, nullptr, hb, NTOK, FFDIM, DMODEL);
    gemm_bf16_wmma<false,true,false><<<gD, blk, 0, stream>>>(
        hb, w2t, b2 + (size_t)i * DMODEL, tmpf, nullptr, NTOK, DMODEL, FFDIM);

    float* dstf = (i == LAYERS - 1) ? out : xf;
    add_layernorm<<<NTOK, blk, 0, stream>>>(curx, tmpf, dstf, xb);
    curx = dstf;
  }
}
